// GLUMLP_18545668784664
// MI455X (gfx1250) — matrix operand order repaired and checked by reading, not compiled, not run on hardware
//
#include <hip/hip_runtime.h>

#define DEV __device__ __forceinline__

constexpr int T = 4096;
constexpr int D = 1024;
constexpr int H = 2048;
constexpr int E = 8;
constexpr int K = 2;
constexpr int P = T * K;      // 8192 token-expert pairs
constexpr int H2 = 2 * H;

typedef __attribute__((ext_vector_type(16))) __bf16 v16bf;
typedef __attribute__((ext_vector_type(8)))  float  v8f;

struct Frag32 { uint4 lo, hi; };
static_assert(sizeof(Frag32) == 32, "frag size");
static_assert(sizeof(v16bf) == 32, "v16bf size");

// ---------------------------------------------------------- conversions ----
DEV unsigned short f2bf(float f) {            // fp32 -> bf16 RNE
  unsigned int u = __float_as_uint(f);
  u += 0x7FFFu + ((u >> 16) & 1u);
  return (unsigned short)(u >> 16);
}

DEV unsigned int f2bf2(float lo, float hi) {  // pack two bf16 into a dword
#if __has_builtin(__builtin_amdgcn_cvt_pk_bf16_f32)
  typedef __attribute__((ext_vector_type(2))) __bf16 v2bf;
  v2bf t = __builtin_amdgcn_cvt_pk_bf16_f32(lo, hi);
  return __builtin_bit_cast(unsigned int, t);
#else
  return (unsigned int)f2bf(lo) | ((unsigned int)f2bf(hi) << 16);
#endif
}

// ------------------------------------------------- async LDS staging -------
// Builtin signature (from hipcc diagnostic): pointers are 16-byte int-vector
// pointers; global operand in AS(1), LDS operand in AS(3).
typedef __attribute__((__vector_size__(4 * sizeof(int)))) int v4i;
typedef __attribute__((address_space(1))) v4i* gptr128;
typedef __attribute__((address_space(3))) v4i* lptr128;

#if __has_builtin(__builtin_amdgcn_global_load_async_to_lds_b128) && \
    __has_builtin(__builtin_amdgcn_s_wait_asynccnt)
#define HAVE_ASYNC_LDS 1
#endif
#if __has_builtin(__builtin_amdgcn_global_store_async_from_lds_b128) && \
    __has_builtin(__builtin_amdgcn_s_wait_asynccnt)
#define HAVE_ASYNC_ST 1
#endif

DEV void stage16(void* lds_dst, const void* glb_src) {   // 16B global -> LDS
#ifdef HAVE_ASYNC_LDS
  __builtin_amdgcn_global_load_async_to_lds_b128(
      (gptr128)(glb_src), (lptr128)(lds_dst), 0, 0);
#else
  *(uint4*)lds_dst = *(const uint4*)glb_src;
#endif
}

DEV void stage_wait() {
#ifdef HAVE_ASYNC_LDS
  __builtin_amdgcn_s_wait_asynccnt(0);
#endif
}

DEV void unstage16(void* glb_dst, const void* lds_src) { // 16B LDS -> global
#ifdef HAVE_ASYNC_ST
  __builtin_amdgcn_global_store_async_from_lds_b128(
      (gptr128)(glb_dst), (lptr128)(lds_src), 0, 0);
#else
  *(uint4*)glb_dst = *(const uint4*)lds_src;
#endif
}

// ------------------------------------------------------ WMMA fragments -----
// A-matrix 16x32 bf16 (ISA 7.12.2): lane half kh holds K = kh*8..+7 and 16+kh*8..+7
DEV v16bf load_frag_a(const unsigned short* rowp, int kh) {
  Frag32 f;
  f.lo = *(const uint4*)(rowp + kh * 8);
  f.hi = *(const uint4*)(rowp + 16 + kh * 8);
  return __builtin_bit_cast(v16bf, f);
}
// B-matrix 32x16 bf16: column n=lane&15, lane half kh holds K = kh*8..+7 and 16+kh*8..+7
DEV v16bf load_frag_b(const unsigned short* colp, int kh) {
  Frag32 f;
  f.lo = *(const uint4*)(colp + kh * 8);
  f.hi = *(const uint4*)(colp + 16 + kh * 8);
  return __builtin_bit_cast(v16bf, f);
}

DEV v8f wmma_bf16(v16bf a, v16bf b, v8f c) {
  return __builtin_amdgcn_wmma_f32_16x16x32_bf16(false, a, false, b, (short)0, c, false, false);
}

// ---------------------------------------------------------------- routing ---
// meta: [0..7]=counts, [8..16]=exclusive offsets
__global__ __launch_bounds__(256) void route_count(const int* __restrict__ idxs,
                                                   int* __restrict__ meta) {
  __shared__ int cnt[E];
  if (threadIdx.x < E) cnt[threadIdx.x] = 0;
  __syncthreads();
  for (int p = threadIdx.x; p < P; p += blockDim.x)
    atomicAdd(&cnt[idxs[p] & (E - 1)], 1);
  __syncthreads();
  if (threadIdx.x == 0) {
    int off = 0;
    for (int e = 0; e < E; ++e) { meta[e] = cnt[e]; meta[E + e] = off; off += cnt[e]; }
    meta[2 * E] = off;
  }
}

// Stable order-preserving compaction per expert -> deterministic slot order.
__global__ __launch_bounds__(256) void route_scatter(const int* __restrict__ idxs,
                                                     const float* __restrict__ pvals,
                                                     const int* __restrict__ meta,
                                                     int* __restrict__ slot_tok,
                                                     float* __restrict__ slot_p) {
  const int e = blockIdx.x;
  __shared__ int wsum[8];
  __shared__ int runBase;
  if (threadIdx.x == 0) runBase = meta[E + e];
  __syncthreads();
  const unsigned lane = threadIdx.x & 31u;
  const int wid = threadIdx.x >> 5;
  for (int base = 0; base < P; base += 256) {
    const int p = base + threadIdx.x;
    const bool flag = ((idxs[p] & (E - 1)) == e);
    unsigned long long m = __ballot(flag);
    const int pre = __popcll(m & ((1ull << lane) - 1ull));
    if (lane == 0) wsum[wid] = __popcll(m);
    __syncthreads();
    int wbase = 0;
    for (int i = 0; i < wid; ++i) wbase += wsum[i];
    if (flag) {
      const int pos = runBase + wbase + pre;
      slot_tok[pos] = p >> 1;          // K == 2
      slot_p[pos]   = pvals[p];
    }
    __syncthreads();
    if (threadIdx.x == 0) {
      int tot = 0;
      for (int i = 0; i < 8; ++i) tot += wsum[i];
      runBase += tot;
    }
    __syncthreads();
  }
}

// ------------------------------------------- weight transpose + bf16 -------
// out[C][R] bf16 = in[R][C] f32, one matrix of R*C per blockIdx.z.
__global__ __launch_bounds__(256) void transpose_convert(const float* __restrict__ in,
                                                         unsigned short* __restrict__ out,
                                                         int Rr, int Cc) {
  const size_t mstride = (size_t)Rr * Cc;
  in  += mstride * blockIdx.z;
  out += mstride * blockIdx.z;
  const int r0 = blockIdx.y * 64, c0 = blockIdx.x * 64;
  __shared__ float tile[64][68];               // [c][r]
  const int tid = threadIdx.x;
  { // coalesced read of 64x64 fp32, store transposed into LDS
    const int row = tid >> 2, cc = (tid & 3) * 16;
    const float* p = in + (size_t)(r0 + row) * Cc + c0 + cc;
    #pragma unroll
    for (int j = 0; j < 16; ++j) tile[cc + j][row] = p[j];
  }
  __syncthreads();
  { // coalesced write of 64 transposed rows, packed to bf16
    const int c = tid >> 2, rr = (tid & 3) * 16;
    unsigned int w[8];
    #pragma unroll
    for (int j = 0; j < 8; ++j) w[j] = f2bf2(tile[c][rr + 2 * j], tile[c][rr + 2 * j + 1]);
    unsigned short* q = out + (size_t)(c0 + c) * Rr + r0 + rr;
    ((uint4*)q)[0] = make_uint4(w[0], w[1], w[2], w[3]);
    ((uint4*)q)[1] = make_uint4(w[4], w[5], w[6], w[7]);
  }
}

// ----------------------------------------------- gather x rows as bf16 -----
__global__ __launch_bounds__(256) void gather_convert_x(const float* __restrict__ x,
                                                        const int* __restrict__ slot_tok,
                                                        unsigned short* __restrict__ xg) {
  const int slot = blockIdx.x;
  const float* src = x + (size_t)slot_tok[slot] * D;
  unsigned short* dst = xg + (size_t)slot * D;
  const int i = threadIdx.x * 4;               // D == 256*4
  float4 v = *(const float4*)(src + i);
  uint2 w;
  w.x = f2bf2(v.x, v.y);
  w.y = f2bf2(v.z, v.w);
  *(uint2*)(dst + i) = w;
}

// ------------------------------------------------- up-proj + SiLU-GLU ------
// Block: 64 slots x 64 cols of BOTH GLU halves; K-stage depth 64.
// 8 waves: w>>2 = h/gates half, w&3 = 16-row strip. act[slot][H] bf16.
__global__ __launch_bounds__(256) void moe_up_glu(const unsigned short* __restrict__ xg,
                                                  const unsigned short* __restrict__ wup_t,
                                                  const int* __restrict__ meta,
                                                  unsigned short* __restrict__ act) {
  const int e   = blockIdx.z;
  const int cnt = meta[e];
  const int mt  = blockIdx.x;
  if (mt * 64 >= cnt) return;
  const int n0       = blockIdx.y * 64;
  const int off      = meta[E + e];
  const int rowsLeft = cnt - mt * 64;

  __shared__ unsigned short As[64][72];        // [slot][k], bf16
  __shared__ unsigned short Bs0[64][72];       // h-half    [n][k]
  __shared__ unsigned short Bs1[64][72];       // gate-half [n][k]
  __shared__ float          Gs[64][66];        // gates exchange
  __shared__ unsigned short Acts[64][64];      // fused bf16 output tile

  const int tid = threadIdx.x;
  const int wid = tid >> 5, lane = tid & 31;
  const int half = wid >> 2, strip = wid & 3;
  const int kh = lane >> 4, ln = lane & 15;

  v8f acc[4];
  #pragma unroll
  for (int i = 0; i < 4; ++i)
    #pragma unroll
    for (int j = 0; j < 8; ++j) acc[i][j] = 0.0f;

  const unsigned short* Wt = wup_t + (size_t)e * H2 * D;   // [h][d] bf16

  for (int k0 = 0; k0 < D; k0 += 64) {
    { // stage A: 64 slots x 64 k (pure bf16 copy)
      const int row = tid >> 2, c0 = (tid & 3) * 16;
      if (mt * 64 + row < cnt) {
        const unsigned short* s = xg + (size_t)(off + mt * 64 + row) * D + k0 + c0;
        stage16(&As[row][c0], s);
        stage16(&As[row][c0 + 8], s + 8);
      } else {
        uint4 z = make_uint4(0u, 0u, 0u, 0u);
        *(uint4*)&As[row][c0] = z;
        *(uint4*)&As[row][c0 + 8] = z;
      }
    }
    { // stage both W_up^T halves: rows are output cols (fragment-major)
      const int n = tid >> 2, c0 = (tid & 3) * 16;
      const unsigned short* s0 = Wt + (size_t)(n0 + n) * D + k0 + c0;
      const unsigned short* s1 = Wt + (size_t)(H + n0 + n) * D + k0 + c0;
      stage16(&Bs0[n][c0], s0); stage16(&Bs0[n][c0 + 8], s0 + 8);
      stage16(&Bs1[n][c0], s1); stage16(&Bs1[n][c0 + 8], s1 + 8);
    }
    stage_wait();
    __syncthreads();

    const unsigned short (*Bsrc)[72] = half ? Bs1 : Bs0;
    #pragma unroll
    for (int kk = 0; kk < 2; ++kk) {
      const v16bf a = load_frag_a(&As[strip * 16 + ln][kk * 32], kh);
      #pragma unroll
      for (int sub = 0; sub < 4; ++sub) {
        const v16bf b = load_frag_b(&Bsrc[sub * 16 + ln][kk * 32], kh);
        acc[sub] = wmma_bf16(a, b, acc[sub]);
      }
    }
    __syncthreads();
  }

  if (half == 1) {  // gates waves publish accumulators
    #pragma unroll
    for (int sub = 0; sub < 4; ++sub)
      #pragma unroll
      for (int r = 0; r < 8; ++r)
        Gs[strip * 16 + kh * 8 + r][sub * 16 + ln] = acc[sub][r];
  }
  __syncthreads();
  if (half == 0) {  // h waves fuse SiLU(g)*h -> bf16 tile
    #pragma unroll
    for (int sub = 0; sub < 4; ++sub)
      #pragma unroll
      for (int r = 0; r < 8; ++r) {
        const int rl = strip * 16 + kh * 8 + r;
        const int c  = sub * 16 + ln;
        const float g = Gs[rl][c];
        const float s = g / (1.0f + __expf(-g));
        Acts[rl][c] = f2bf(s * acc[sub][r]);
      }
  }
  __syncthreads();

  { // tile writeback, LDS -> global (async store path if available)
    const int row = tid >> 2, c0 = (tid & 3) * 16;
    if (row < rowsLeft) {
      unsigned short* dst = act + (size_t)(off + mt * 64 + row) * H + n0 + c0;
      unstage16(dst, &Acts[row][c0]);
      unstage16(dst + 8, &Acts[row][c0 + 8]);
    }
  }
}

// ----------------------------------------------- down-proj + scatter-add ---
// Block: 64 slots x 128 d-cols; K-stage depth 64. w&3 = row strip, w>>2 = col half.
__global__ __launch_bounds__(256) void moe_down(const unsigned short* __restrict__ act,
                                                const unsigned short* __restrict__ wdn_t,
                                                const int* __restrict__ meta,
                                                const int* __restrict__ slot_tok,
                                                const float* __restrict__ slot_p,
                                                float* __restrict__ out) {
  const int e   = blockIdx.z;
  const int cnt = meta[e];
  const int mt  = blockIdx.x;
  if (mt * 64 >= cnt) return;
  const int d0       = blockIdx.y * 128;
  const int off      = meta[E + e];
  const int rowsLeft = cnt - mt * 64;

  __shared__ unsigned short As[64][72];        // act tile [slot][k]
  __shared__ unsigned short Bst[128][72];      // W_down^T [d][k]
  __shared__ int   toks[64];
  __shared__ float ps[64];

  const int tid = threadIdx.x;
  if (tid < 64) {
    const bool valid = (tid < rowsLeft);
    toks[tid] = valid ? slot_tok[off + mt * 64 + tid] : 0;
    ps[tid]   = valid ? slot_p[off + mt * 64 + tid] : 0.0f;
  }

  const int wid = tid >> 5, lane = tid & 31;
  const int ch = wid >> 2, strip = wid & 3;
  const int kh = lane >> 4, ln = lane & 15;

  v8f acc[4];
  #pragma unroll
  for (int i = 0; i < 4; ++i)
    #pragma unroll
    for (int j = 0; j < 8; ++j) acc[i][j] = 0.0f;

  const unsigned short* Wt = wdn_t + (size_t)e * D * H;    // [d][h] bf16

  for (int k0 = 0; k0 < H; k0 += 64) {
    { // stage A from act (bf16 copy)
      const int row = tid >> 2, c0 = (tid & 3) * 16;
      if (mt * 64 + row < cnt) {
        const unsigned short* s = act + (size_t)(off + mt * 64 + row) * H + k0 + c0;
        stage16(&As[row][c0], s);
        stage16(&As[row][c0 + 8], s + 8);
      } else {
        uint4 z = make_uint4(0u, 0u, 0u, 0u);
        *(uint4*)&As[row][c0] = z;
        *(uint4*)&As[row][c0 + 8] = z;
      }
    }
    { // stage B: 128 d-rows x 64 k
      const int d = tid >> 1, c0 = (tid & 1) * 32;
      const unsigned short* s = Wt + (size_t)(d0 + d) * H + k0 + c0;
      stage16(&Bst[d][c0],      s);
      stage16(&Bst[d][c0 + 8],  s + 8);
      stage16(&Bst[d][c0 + 16], s + 16);
      stage16(&Bst[d][c0 + 24], s + 24);
    }
    stage_wait();
    __syncthreads();

    #pragma unroll
    for (int kk = 0; kk < 2; ++kk) {
      const v16bf a = load_frag_a(&As[strip * 16 + ln][kk * 32], kh);
      #pragma unroll
      for (int sub = 0; sub < 4; ++sub) {
        const v16bf b = load_frag_b(&Bst[ch * 64 + sub * 16 + ln][kk * 32], kh);
        acc[sub] = wmma_bf16(a, b, acc[sub]);
      }
    }
    __syncthreads();
  }

  // scale by gate prob; exactly K=2 float contributions/element -> the two-way
  // atomic sum is order-independent -> deterministic.
  #pragma unroll
  for (int sub = 0; sub < 4; ++sub) {
    const int c = d0 + ch * 64 + sub * 16 + ln;
    #pragma unroll
    for (int r = 0; r < 8; ++r) {
      const int rl = strip * 16 + kh * 8 + r;
      if (rl < rowsLeft)
        unsafeAtomicAdd(out + (size_t)toks[rl] * D + c, ps[rl] * acc[sub][r]);
    }
  }
}

// ---------------------------------------------------------------- launch ----
extern "C" void kernel_launch(void* const* d_in, const int* in_sizes, int n_in,
                              void* d_out, int out_size, void* d_ws, size_t ws_size,
                              hipStream_t stream) {
  const float* x    = (const float*)d_in[0];
  const float* ep   = (const float*)d_in[1];
  const int*   eidx = (const int*)d_in[2];
  const float* wu   = (const float*)d_in[3];
  const float* wd   = (const float*)d_in[4];
  float* out = (float*)d_out;

  // workspace layout (bytes)
  char* ws = (char*)d_ws;
  size_t o = 0;
  int*   meta = (int*)(ws + o);              o += 256;
  int*   stok = (int*)(ws + o);              o += (size_t)P * sizeof(int);
  float* sp   = (float*)(ws + o);            o += (size_t)P * sizeof(float);
  unsigned short* xg    = (unsigned short*)(ws + o); o += (size_t)P * D * 2;   // 16.8 MB
  unsigned short* actb  = (unsigned short*)(ws + o); o += (size_t)P * H * 2;   // 33.6 MB
  unsigned short* wup_t = (unsigned short*)(ws + o); o += (size_t)E * H2 * D * 2; // 67 MB
  unsigned short* wdn_t = (unsigned short*)(ws + o); o += (size_t)E * D * H * 2;  // 33.6 MB

  (void)hipMemsetAsync(out, 0, sizeof(float) * (size_t)T * D, stream);

  route_count  <<<1, 256, 0, stream>>>(eidx, meta);
  route_scatter<<<E, 256, 0, stream>>>(eidx, ep, meta, stok, sp);

  // W_up [E][D][2H] f32 -> wup_t [E][2H][D] bf16 ; W_down [E][H][D] -> wdn_t [E][D][H]
  transpose_convert<<<dim3(H2 / 64, D / 64, E), 256, 0, stream>>>(wu, wup_t, D, H2);
  transpose_convert<<<dim3(D / 64, H / 64, E), 256, 0, stream>>>(wd, wdn_t, H, D);
  gather_convert_x <<<P, 256, 0, stream>>>(x, stok, xg);

  moe_up_glu<<<dim3(P / 64, H / 64, E), 256, 0, stream>>>(xg, wup_t, meta, actb);
  moe_down  <<<dim3(P / 64, D / 128, E), 256, 0, stream>>>(actb, wdn_t, meta, stok, sp, out);
}
